// AdaptiveLSTMCellWithRes_19473381720316
// MI455X (gfx1250) — matrix operand order repaired and checked by reading, not compiled, not run on hardware
//
#include <hip/hip_runtime.h>

// ---------------------------------------------------------------------------
// MI455X (gfx1250) AdaptiveLSTMCellWithRes
// Roofline: ~258 GFLOP of GEMM vs ~230MB HBM traffic -> compute bound.
// Strategy: bf16 WMMA (f32 accumulate) for all GEMMs; gates fused into one
// M=8192 x N=6144 x K=2048 GEMM over packed [x|h] input. Data movement uses
// gfx1250 async global->LDS copies, triple-buffered (2 stages in flight,
// s_wait_asynccnt 4) so each HBM copy gets ~2 compute phases of slack.
// ---------------------------------------------------------------------------

typedef __bf16 bf16;
typedef __attribute__((ext_vector_type(16))) __bf16 v16bf;
typedef __attribute__((ext_vector_type(8)))  __bf16 v8bf;
typedef __attribute__((ext_vector_type(8)))  float  v8f;
typedef int v4i __attribute__((__vector_size__(16)));   // matches builtin param type

union FragBF { v16bf v; v8bf h[2]; };

#define BM 128
#define BN 128
#define BKK 32
#define LDS_STRIDE 40   // bf16 elems/row: 32 data + 8 pad = 80B (16B aligned, conflict-free)

enum { ACT_NONE = 0, ACT_RELU = 1, ACT_GATES = 2 };

// --- gfx1250 feature probes --------------------------------------------------
#if defined(__has_builtin)
#  if __has_builtin(__builtin_amdgcn_global_load_async_to_lds_b128)
#    define HAVE_ASYNC_COPY 1
#  endif
#endif
#ifndef HAVE_ASYNC_COPY
#  define HAVE_ASYNC_COPY 0
#endif

#if defined(__has_builtin) && __has_builtin(__builtin_amdgcn_s_wait_asynccnt)
#  define ASYNC_WAIT(N) __builtin_amdgcn_s_wait_asynccnt(N)
#else
#  define ASYNC_WAIT(N) asm volatile("s_wait_asynccnt %0" :: "i"(N) : "memory")
#endif

__device__ __forceinline__ float ftanh(float x) {
#if defined(__has_builtin) && __has_builtin(__builtin_amdgcn_tanhf)
    return __builtin_amdgcn_tanhf(x);
#elif defined(__has_builtin) && __has_builtin(__builtin_amdgcn_tanh_f32)
    return __builtin_amdgcn_tanh_f32(x);
#else
    return tanhf(x);
#endif
}

__device__ __forceinline__ float fsigmoid(float x) { return 1.0f / (1.0f + __expf(-x)); }

// ---------------------------------------------------------------------------
// bf16 WMMA GEMM: C[M,N] = act(A[M,K] @ B[N,K]^T + bias[N])
// 256 threads = 8 waves; block tile 128x128, K-step 32.
// Wave grid 2(M) x 4(N); each wave computes 64x32 = 4x2 WMMA tiles.
// Requires: N block-aligned so gate id (col>>10) is block-uniform for ACT_GATES.
// ---------------------------------------------------------------------------
template <int ACT, bool OUT_BF16>
__global__ __launch_bounds__(256)
void gemm_bf16_wmma(const bf16* __restrict__ A, int lda,
                    const bf16* __restrict__ Bw, int ldb,
                    const float* __restrict__ bias,
                    float* __restrict__ Cf, bf16* __restrict__ Cb, int ldc,
                    int K)
{
#if HAVE_ASYNC_COPY
    __shared__ bf16 shA[3][BM * LDS_STRIDE];
    __shared__ bf16 shB[3][BN * LDS_STRIDE];
#else
    __shared__ bf16 shA[2][BM * LDS_STRIDE];
    __shared__ bf16 shB[2][BN * LDS_STRIDE];
#endif

    const int tid  = threadIdx.x;
    const int lane = tid & 31;
    const int wave = tid >> 5;
    const int wm   = wave & 1;    // wave M index (0..1)
    const int wn   = wave >> 1;   // wave N index (0..3)
    const int hlf  = lane >> 4;   // lane half (0..1)
    const int r16  = lane & 15;

    const int bm = blockIdx.y * BM;
    const int bn = blockIdx.x * BN;

    // Global->LDS staging mapping: thread covers 2 rows of A and 2 rows of B,
    // 16B (8 bf16) per row chunk.
    const int lr = tid >> 2;          // 0..63
    const int lc = (tid & 3) * 8;     // 0,8,16,24 bf16 elems

    const bf16* gA0 = A  + (size_t)(bm + lr)      * lda + lc;
    const bf16* gA1 = A  + (size_t)(bm + lr + 64) * lda + lc;
    const bf16* gB0 = Bw + (size_t)(bn + lr)      * ldb + lc;
    const bf16* gB1 = Bw + (size_t)(bn + lr + 64) * ldb + lc;

    v8f acc[4][2] = {};

    const int kTiles = K / BKK;

#if HAVE_ASYNC_COPY
    // ---- gfx1250 async copy: global -> LDS without VGPR staging ----
    auto issue_copy = [&](int st, int k0) {
        __builtin_amdgcn_global_load_async_to_lds_b128(
            (__attribute__((address_space(1))) v4i*)(gA0 + k0),
            (__attribute__((address_space(3))) v4i*)&shA[st][(lr)      * LDS_STRIDE + lc], 0, 0);
        __builtin_amdgcn_global_load_async_to_lds_b128(
            (__attribute__((address_space(1))) v4i*)(gA1 + k0),
            (__attribute__((address_space(3))) v4i*)&shA[st][(lr + 64) * LDS_STRIDE + lc], 0, 0);
        __builtin_amdgcn_global_load_async_to_lds_b128(
            (__attribute__((address_space(1))) v4i*)(gB0 + k0),
            (__attribute__((address_space(3))) v4i*)&shB[st][(lr)      * LDS_STRIDE + lc], 0, 0);
        __builtin_amdgcn_global_load_async_to_lds_b128(
            (__attribute__((address_space(1))) v4i*)(gB1 + k0),
            (__attribute__((address_space(3))) v4i*)&shB[st][(lr + 64) * LDS_STRIDE + lc], 0, 0);
    };
#else
    uint4 ra0, ra1, rb0, rb1;
    auto gfetch = [&](int k0) {
        ra0 = *(const uint4*)(gA0 + k0);
        ra1 = *(const uint4*)(gA1 + k0);
        rb0 = *(const uint4*)(gB0 + k0);
        rb1 = *(const uint4*)(gB1 + k0);
    };
    auto lstore = [&](int st) {
        *(uint4*)&shA[st][(lr)      * LDS_STRIDE + lc] = ra0;
        *(uint4*)&shA[st][(lr + 64) * LDS_STRIDE + lc] = ra1;
        *(uint4*)&shB[st][(lr)      * LDS_STRIDE + lc] = rb0;
        *(uint4*)&shB[st][(lr + 64) * LDS_STRIDE + lc] = rb1;
    };
#endif

    auto compute = [&](int cur) {
        // A fragments (16-bit A 16x32 layout: lanes 0-15 hold K 0-7 & 16-23,
        // lanes 16-31 hold K 8-15 & 24-31).
        FragBF af[4];
#pragma unroll
        for (int mi = 0; mi < 4; ++mi) {
            const bf16* p = &shA[cur][(wm * 64 + mi * 16 + r16) * LDS_STRIDE + hlf * 8];
            af[mi].h[0] = *(const v8bf*)p;
            af[mi].h[1] = *(const v8bf*)(p + 16);
        }
        // B fragments (B 32x16: lane = column; lanes 0-15 hold K 0-7 & 16-23,
        // lanes 16-31 hold K 8-15 & 24-31).
        FragBF bfr[2];
#pragma unroll
        for (int ni = 0; ni < 2; ++ni) {
            const bf16* p = &shB[cur][(wn * 32 + ni * 16 + r16) * LDS_STRIDE + hlf * 8];
            bfr[ni].h[0] = *(const v8bf*)p;
            bfr[ni].h[1] = *(const v8bf*)(p + 16);
        }
#pragma unroll
        for (int mi = 0; mi < 4; ++mi)
#pragma unroll
            for (int ni = 0; ni < 2; ++ni)
                acc[mi][ni] = __builtin_amdgcn_wmma_f32_16x16x32_bf16(
                    false, af[mi].v, false, bfr[ni].v, (short)0, acc[mi][ni],
                    false, false);
    };

#if HAVE_ASYNC_COPY
    // Triple-buffered pipeline, 2 stages in flight. Async loads complete in
    // order, so s_wait_asynccnt 4 guarantees the oldest stage (4 copies) has
    // landed while the newest 4 remain outstanding.
    issue_copy(0, 0);
    if (kTiles > 1) issue_copy(1, BKK);
    int bufC = 0;   // buffer holding stage kt
    int bufI = 2;   // buffer for stage kt+2
    for (int kt = 0; kt < kTiles; ++kt) {
        if (kt + 1 < kTiles) ASYNC_WAIT(4);   // stage kt done, kt+1 may fly
        else                 ASYNC_WAIT(0);
        __syncthreads();                      // all waves see stage kt in LDS;
                                              // also: all waves done with stage kt-1
        if (kt + 2 < kTiles) {
            issue_copy(bufI, (kt + 2) * BKK); // reuses stage kt-1's buffer (safe)
            __builtin_prefetch(gA0 + (kt + 3 < kTiles ? kt + 3 : kt + 2) * BKK, 0, 0);
        }
        compute(bufC);
        bufC = (bufC == 2) ? 0 : bufC + 1;
        bufI = (bufI == 2) ? 0 : bufI + 1;
        __syncthreads();                      // don't let fast waves re-wait early
    }
#else
    gfetch(0);
    lstore(0);
    __syncthreads();
    for (int kt = 0; kt < kTiles; ++kt) {
        const int cur = kt & 1;
        if (kt + 1 < kTiles) gfetch((kt + 1) * BKK);
        if (kt + 2 < kTiles) {
            __builtin_prefetch(gA0 + (kt + 2) * BKK, 0, 0);
            __builtin_prefetch(gA1 + (kt + 2) * BKK, 0, 0);
        }
        compute(cur);
        if (kt + 1 < kTiles) lstore(cur ^ 1);
        __syncthreads();
    }
#endif

    // Epilogue. C layout: VGPR v -> lanes 0-15: M = v, lanes 16-31: M = 8+v.
    // Activation for ACT_GATES is block-uniform: gate = bn >> 10 (BN | 1024).
    const int gate = bn >> 10;
#pragma unroll
    for (int ni = 0; ni < 2; ++ni) {
        const int col = bn + wn * 32 + ni * 16 + r16;
        const float bs = bias[col];
#pragma unroll
        for (int mi = 0; mi < 4; ++mi) {
#pragma unroll
            for (int v = 0; v < 8; ++v) {
                const int row = bm + wm * 64 + mi * 16 + hlf * 8 + v;
                float val = acc[mi][ni][v] + bs;
                if (ACT == ACT_RELU) {
                    val = fmaxf(val, 0.0f);
                } else if (ACT == ACT_GATES) {
                    if (gate == 3)      val = ftanh(val);        // c_hat
                    else if (gate == 5) val = fmaxf(val, 0.0f);  // a1 relu
                    else                val = fsigmoid(val);     // i,f,o,s
                }
                if (OUT_BF16) Cb[(size_t)row * ldc + col] = (bf16)val;
                else          Cf[(size_t)row * ldc + col] = val;
            }
        }
    }
}

// ---------------------------------------------------------------------------
// Packing / conversion kernels
// ---------------------------------------------------------------------------
__global__ __launch_bounds__(256)
void pack_xh(const float* __restrict__ x, const float* __restrict__ h,
             bf16* __restrict__ XH)
{
    const size_t idx = (size_t)blockIdx.x * 256 + threadIdx.x;  // over 8192*2048
    const int b = (int)(idx >> 11);
    const int k = (int)(idx & 2047);
    const float v = (k < 1024) ? x[(size_t)b * 1024 + k]
                               : h[(size_t)b * 1024 + (k - 1024)];
    XH[idx] = (bf16)v;
}

__global__ __launch_bounds__(256)
void pack_gates_w(const float* __restrict__ Wi, const float* __restrict__ Wf,
                  const float* __restrict__ Wo, const float* __restrict__ Wc,
                  const float* __restrict__ Ws,
                  const float* __restrict__ Ui, const float* __restrict__ Uf,
                  const float* __restrict__ Uo, const float* __restrict__ Uc,
                  const float* __restrict__ Us,
                  const float* __restrict__ a1w, bf16* __restrict__ Wg)
{
    const size_t idx = (size_t)blockIdx.x * 256 + threadIdx.x;  // over 6144*2048
    const int n = (int)(idx >> 11);
    const int k = (int)(idx & 2047);
    const int g = n >> 10;
    const int nn = n & 1023;
    float v;
    if (g == 5) {
        v = a1w[(size_t)nn * 2048 + k];
    } else {
        const float* W = (g == 0) ? Wi : (g == 1) ? Wf : (g == 2) ? Wo : (g == 3) ? Wc : Ws;
        const float* U = (g == 0) ? Ui : (g == 1) ? Uf : (g == 2) ? Uo : (g == 3) ? Uc : Us;
        v = (k < 1024) ? W[(size_t)nn * 1024 + k] : U[(size_t)nn * 1024 + (k - 1024)];
    }
    Wg[idx] = (bf16)v;
}

__global__ __launch_bounds__(256)
void pack_gates_bias(const float* __restrict__ bWi, const float* __restrict__ bWf,
                     const float* __restrict__ bWo, const float* __restrict__ bWc,
                     const float* __restrict__ bWs,
                     const float* __restrict__ bUi, const float* __restrict__ bUf,
                     const float* __restrict__ bUo, const float* __restrict__ bUc,
                     const float* __restrict__ bUs,
                     const float* __restrict__ a1b, float* __restrict__ bg)
{
    const int idx = blockIdx.x * 256 + threadIdx.x;  // over 6144
    const int g = idx >> 10;
    const int nn = idx & 1023;
    float v;
    if (g == 5) {
        v = a1b[nn];
    } else {
        const float* bw = (g == 0) ? bWi : (g == 1) ? bWf : (g == 2) ? bWo : (g == 3) ? bWc : bWs;
        const float* bu = (g == 0) ? bUi : (g == 1) ? bUf : (g == 2) ? bUo : (g == 3) ? bUc : bUs;
        v = bw[nn] + bu[nn];
    }
    bg[idx] = v;
}

__global__ __launch_bounds__(256)
void cvt_f32_bf16(const float* __restrict__ s, bf16* __restrict__ d, int n)
{
    const int i = blockIdx.x * 256 + threadIdx.x;
    if (i < n) d[i] = (bf16)s[i];
}

// alpha = sigmoid(relu_a1 @ a2_w^T + a2_b) : N=1 -> one wave per row reduction
__global__ __launch_bounds__(256)
void alpha_kernel(const float* __restrict__ G,   // [B, 6144]; a1 cols at 5120..6143
                  const float* __restrict__ a2w, const float* __restrict__ a2b,
                  float* __restrict__ alpha)
{
    const int wave = threadIdx.x >> 5;
    const int lane = threadIdx.x & 31;
    const int row  = blockIdx.x * 8 + wave;
    const float* g = G + (size_t)row * 6144 + 5120;
    float s = 0.0f;
#pragma unroll 4
    for (int k = lane; k < 1024; k += 32) s += g[k] * a2w[k];
#pragma unroll
    for (int off = 16; off > 0; off >>= 1) s += __shfl_xor(s, off, 32);
    if (lane == 0) alpha[row] = fsigmoid(s + a2b[0]);
}

// c_t = f*c_prev + i*c_hat*s*alpha + residual ; h_t = o*tanh(c_t)
__global__ __launch_bounds__(256)
void finalize_kernel(const float* __restrict__ G, const float* __restrict__ alpha,
                     const float* __restrict__ R, const float* __restrict__ c_prev,
                     float* __restrict__ out)
{
    const size_t idx = (size_t)blockIdx.x * 256 + threadIdx.x;  // over 8192*1024
    const int b = (int)(idx >> 10);
    const int j = (int)(idx & 1023);
    const float* g = G + (size_t)b * 6144;
    const float i_t  = g[j];
    const float f_t  = g[1024 + j];
    const float o_t  = g[2048 + j];
    const float ch_t = g[3072 + j];
    const float s_t  = g[4096 + j];
    const float ct = f_t * c_prev[idx] + i_t * ch_t * s_t * alpha[b] + R[idx];
    const float ht = o_t * ftanh(ct);
    out[idx] = ht;                              // h_t
    out[(size_t)8192 * 1024 + idx] = ct;        // c_t
}

// ---------------------------------------------------------------------------
// Launcher
// ---------------------------------------------------------------------------
extern "C" void kernel_launch(void* const* d_in, const int* in_sizes, int n_in,
                              void* d_out, int out_size, void* d_ws, size_t ws_size,
                              hipStream_t stream)
{
    constexpr int kB = 8192;
    constexpr int kD = 1024;

    const float* x      = (const float*)d_in[0];
    const float* h_prev = (const float*)d_in[1];
    const float* c_prev = (const float*)d_in[2];
    const float* Wi_w = (const float*)d_in[3],  *Wi_b = (const float*)d_in[4];
    const float* Wf_w = (const float*)d_in[5],  *Wf_b = (const float*)d_in[6];
    const float* Wo_w = (const float*)d_in[7],  *Wo_b = (const float*)d_in[8];
    const float* Wc_w = (const float*)d_in[9],  *Wc_b = (const float*)d_in[10];
    const float* Ws_w = (const float*)d_in[11], *Ws_b = (const float*)d_in[12];
    const float* Ui_w = (const float*)d_in[13], *Ui_b = (const float*)d_in[14];
    const float* Uf_w = (const float*)d_in[15], *Uf_b = (const float*)d_in[16];
    const float* Uo_w = (const float*)d_in[17], *Uo_b = (const float*)d_in[18];
    const float* Uc_w = (const float*)d_in[19], *Uc_b = (const float*)d_in[20];
    const float* Us_w = (const float*)d_in[21], *Us_b = (const float*)d_in[22];
    const float* a1_w = (const float*)d_in[23], *a1_b = (const float*)d_in[24];
    const float* a2_w = (const float*)d_in[25], *a2_b = (const float*)d_in[26];
    const float* r1_w = (const float*)d_in[27], *r1_b = (const float*)d_in[28];
    const float* r2_w = (const float*)d_in[29], *r2_b = (const float*)d_in[30];
    const float* r3_w = (const float*)d_in[31], *r3_b = (const float*)d_in[32];

    size_t off = 0;
    auto take = [&](size_t bytes) -> void* {
        void* p = (char*)d_ws + off;
        off += (bytes + 255) & ~(size_t)255;
        return p;
    };
    bf16*  XH    = (bf16*)take((size_t)kB * 2048 * sizeof(bf16));
    bf16*  Wg    = (bf16*)take((size_t)6144 * 2048 * sizeof(bf16));
    float* bg    = (float*)take((size_t)6144 * sizeof(float));
    bf16*  Wr1   = (bf16*)take((size_t)kD * kD * sizeof(bf16));
    bf16*  Wr2   = (bf16*)take((size_t)kD * kD * sizeof(bf16));
    bf16*  Wr3   = (bf16*)take((size_t)kD * kD * sizeof(bf16));
    float* G     = (float*)take((size_t)kB * 6144 * sizeof(float));
    float* alpha = (float*)take((size_t)kB * sizeof(float));
    bf16*  T1    = (bf16*)take((size_t)kB * kD * sizeof(bf16));
    bf16*  T2    = (bf16*)take((size_t)kB * kD * sizeof(bf16));
    float* Rres  = (float*)take((size_t)kB * kD * sizeof(float));

    // 1) Pack inputs and weights to bf16
    pack_xh<<<(kB * 2048) / 256, 256, 0, stream>>>(x, h_prev, XH);
    pack_gates_w<<<(6144 * 2048) / 256, 256, 0, stream>>>(
        Wi_w, Wf_w, Wo_w, Wc_w, Ws_w, Ui_w, Uf_w, Uo_w, Uc_w, Us_w, a1_w, Wg);
    pack_gates_bias<<<6144 / 256, 256, 0, stream>>>(
        Wi_b, Wf_b, Wo_b, Wc_b, Ws_b, Ui_b, Uf_b, Uo_b, Uc_b, Us_b, a1_b, bg);
    cvt_f32_bf16<<<(kD * kD) / 256, 256, 0, stream>>>(r1_w, Wr1, kD * kD);
    cvt_f32_bf16<<<(kD * kD) / 256, 256, 0, stream>>>(r2_w, Wr2, kD * kD);
    cvt_f32_bf16<<<(kD * kD) / 256, 256, 0, stream>>>(r3_w, Wr3, kD * kD);

    // 2) Fused gates GEMM: [8192,2048] x [6144,2048]^T -> [8192,6144] (+act)
    {
        dim3 grid(6144 / BN, kB / BM);
        gemm_bf16_wmma<ACT_GATES, false><<<grid, 256, 0, stream>>>(
            XH, 2048, Wg, 2048, bg, G, nullptr, 6144, 2048);
    }

    // 3) alpha = sigmoid(a2(relu_a1))
    alpha_kernel<<<kB / 8, 256, 0, stream>>>(G, a2_w, a2_b, alpha);

    // 4) Residual chain: r1(h)->relu->r2->relu->r3
    {
        dim3 grid(kD / BN, kB / BM);
        // r1: input = h half of XH (lda=2048, offset 1024)
        gemm_bf16_wmma<ACT_RELU, true><<<grid, 256, 0, stream>>>(
            XH + 1024, 2048, Wr1, kD, r1_b, nullptr, T1, kD, kD);
        gemm_bf16_wmma<ACT_RELU, true><<<grid, 256, 0, stream>>>(
            T1, kD, Wr2, kD, r2_b, nullptr, T2, kD, kD);
        gemm_bf16_wmma<ACT_NONE, false><<<grid, 256, 0, stream>>>(
            T2, kD, Wr3, kD, r3_b, Rres, nullptr, kD, kD);
    }

    // 5) Elementwise finalize -> d_out = [h_t | c_t]
    finalize_kernel<<<(kB * kD) / 256, 256, 0, stream>>>(
        G, alpha, Rres, c_prev, (float*)d_out);
}
